// Attention_81269371175200
// MI455X (gfx1250) — matrix operand order repaired and checked by reading, not compiled, not run on hardware
//
#include <hip/hip_runtime.h>
#include <hip/hip_bf16.h>

typedef __attribute__((ext_vector_type(16))) __bf16 v16bf;
typedef __attribute__((ext_vector_type(8)))  __bf16 v8bf;
typedef __attribute__((ext_vector_type(8)))  float  v8f;
typedef __attribute__((ext_vector_type(4)))  float  v4f;

#define BB 32
#define SS 2048
#define KDIM 1024   // 2*ENC_H
#define NDIM 512    // DEC_H
#define TM 64       // rows per workgroup in the score GEMM

// branch-free tanh: 1 - 2/(exp(2x)+1); v_exp_f32 + v_rcp_f32, no divergence
__device__ __forceinline__ float fast_tanh(float x) {
  float ex = __expf(x + x);
  return 1.0f - 2.0f * __builtin_amdgcn_rcpf(ex + 1.0f);
}

// ---------------- kernel 1: hb[b,d] = hidden[b,:] . attn_w[d,0:512] + attn_b[d]
__global__ void k_prep_hb(const float* __restrict__ hidden,
                          const float* __restrict__ attn_w,
                          const float* __restrict__ attn_b,
                          float* __restrict__ hb) {
  int idx = blockIdx.x * blockDim.x + threadIdx.x;   // 32*512 = 16384
  int b = idx >> 9, d = idx & 511;
  const float* hrow = hidden + b * NDIM;
  const float* wrow = attn_w + (size_t)d * (NDIM + KDIM);
  float s = 0.f;
#pragma unroll 8
  for (int k = 0; k < NDIM; ++k) s += hrow[k] * wrow[k];
  hb[idx] = s + attn_b[d];
}

// ---------------- kernel 2: W_e -> bf16, row-major [512][1024]
__global__ void k_prep_wbf(const float* __restrict__ attn_w, __bf16* __restrict__ wbf) {
  int i = (blockIdx.x * blockDim.x + threadIdx.x) * 4;  // over 512*1024
  int d = i >> 10, k = i & 1023;
  const v4f f = *(const v4f*)(attn_w + (size_t)d * (NDIM + KDIM) + NDIM + k);
  union { ushort4 u; __bf16 h[4]; } pk;
  pk.h[0] = (__bf16)f[0]; pk.h[1] = (__bf16)f[1];
  pk.h[2] = (__bf16)f[2]; pk.h[3] = (__bf16)f[3];
  *(ushort4*)(wbf + i) = pk.u;
}

// ---------------- kernel 3: fused GEMM(bf16 WMMA) + tanh + dot(v) -> scores[B*S]
__launch_bounds__(128)
__global__ void k_scores(const float* __restrict__ enc,    // [B*S,1024] f32
                         const __bf16* __restrict__ wbf,   // [512,1024] bf16
                         const float* __restrict__ hb,     // [B,512]
                         const float* __restrict__ v_w,    // [512]
                         const float* __restrict__ v_b,    // [1]
                         float* __restrict__ scores) {     // [B*S]
  __shared__ __align__(64) __bf16 As[TM * KDIM];   // 131072 B
  __shared__ __align__(64) __bf16 Bs[16 * KDIM];   //  32768 B
  const int tid   = threadIdx.x;
  const int wave  = tid >> 5;
  const int lane  = tid & 31;
  const int lhalf = lane >> 4;      // 0 or 1
  const int l16   = lane & 15;
  const long long r0 = (long long)blockIdx.x * TM; // flattened b*S+s base
  const int b = (int)(r0 >> 11);    // / 2048 (TM divides S -> tile within one batch)

  // ---- stage A tile: 64 rows x 1024 f32 -> bf16 in LDS (coalesced v4f loads)
  for (int it = 0; it < (TM * KDIM / 4) / 128; ++it) {
    int u   = it * 128 + tid;       // float4 index
    int row = u >> 8;               // / 256
    int c4  = u & 255;
    v4f f = __builtin_nontemporal_load(
        (const v4f*)(enc + (r0 + row) * (long long)KDIM + c4 * 4));
    union { ushort4 u4; __bf16 h[4]; } pk;
    pk.h[0] = (__bf16)f[0]; pk.h[1] = (__bf16)f[1];
    pk.h[2] = (__bf16)f[2]; pk.h[3] = (__bf16)f[3];
    *(ushort4*)(As + row * KDIM + c4 * 4) = pk.u4;
  }
  __syncthreads();

  const int arow = wave * 16 + l16;
  float sacc[8];
#pragma unroll
  for (int j = 0; j < 8; ++j) sacc[j] = 0.f;

  for (int nc = 0; nc < 32; ++nc) {
    __syncthreads();  // protect Bs reuse
    // ---- stage B chunk: W_e rows [nc*16, nc*16+16) : 32 KB, shared by all 4 waves
    for (int it = 0; it < 16; ++it) {
      int u = it * 128 + tid;   // uint4 index over 32768 B
      ((uint4*)Bs)[u] = ((const uint4*)(wbf + (size_t)(nc * 16) * KDIM))[u];
    }
    __syncthreads();

    const int n    = nc * 16 + l16;
    const float hbv = hb[b * NDIM + n];
    const float vv  = v_w[n];

    v8f acc = {};
#pragma unroll 4
    for (int kt = 0; kt < 32; ++kt) {
      const int k0  = kt * 32;
      // A operand (this code's contiguous k order, same as B): lane = row M, 16 contiguous K per lane (lanes 16-31 +16)
      v16bf a  = *(const v16bf*)(As + arow * KDIM + k0 + (lhalf << 4));
      // B operand (same contiguous k order as A): lane = column N, 16 contiguous K per lane
      v16bf bm = *(const v16bf*)(Bs + l16 * KDIM + k0 + (lhalf << 4));
      acc = __builtin_amdgcn_wmma_f32_16x16x32_bf16(
              false, a, false, bm, (short)0, acc, false, false);
    }
    // C layout: VGPR j, lanes 0-15 -> M=j, lanes 16-31 -> M=8+j, N=lane&15
#pragma unroll
    for (int j = 0; j < 8; ++j) {
      float t = fast_tanh(acc[j] + hbv) * vv;
      t += __shfl_xor(t, 8, 32);
      t += __shfl_xor(t, 4, 32);
      t += __shfl_xor(t, 2, 32);
      t += __shfl_xor(t, 1, 32);
      sacc[j] += t;           // meaningful in lanes 0 and 16
    }
  }
  if (l16 == 0) {
    const float vb = v_b[0];
#pragma unroll
    for (int j = 0; j < 8; ++j) {
      long long r = r0 + wave * 16 + lhalf * 8 + j;
      scores[r] = sacc[j] + vb;
    }
  }
}

// ---------------- kernel 4: masked softmax over S per batch row
__global__ void k_softmax(const float* __restrict__ scores,
                          const int* __restrict__ lengths,
                          float* __restrict__ energy) {
  __shared__ float red[256];
  const int b = blockIdx.x, tid = threadIdx.x;
  const int len = lengths[b];
  const float* sr = scores + b * SS;
  float mx = -INFINITY;
  for (int s = tid; s < SS; s += 256) if (s < len) mx = fmaxf(mx, sr[s]);
  red[tid] = mx; __syncthreads();
  for (int o = 128; o > 0; o >>= 1) { if (tid < o) red[tid] = fmaxf(red[tid], red[tid + o]); __syncthreads(); }
  mx = red[0]; __syncthreads();
  float sum = 0.f;
  for (int s = tid; s < SS; s += 256) if (s < len) sum += __expf(sr[s] - mx);
  red[tid] = sum; __syncthreads();
  for (int o = 128; o > 0; o >>= 1) { if (tid < o) red[tid] += red[tid + o]; __syncthreads(); }
  const float inv = 1.0f / red[0];
  for (int s = tid; s < SS; s += 256)
    energy[b * SS + s] = (s < len) ? __expf(sr[s] - mx) * inv : 0.f;
}

// ---------------- kernel 5: split-S partial weighted sums (deterministic, no atomics)
__global__ void k_ctx_part(const float* __restrict__ enc,
                           const float* __restrict__ energy,
                           float* __restrict__ part) {
  // grid (4 e-chunks, 16 s-chunks, 32 b), block 256
  const int e  = blockIdx.x * 256 + threadIdx.x;
  const int sc = blockIdx.y;
  const int b  = blockIdx.z;
  const float* eb   = energy + b * SS + sc * 128;
  const float* base = enc + ((long long)b * SS + sc * 128) * KDIM + e;
  float acc = 0.f;
#pragma unroll 4
  for (int s = 0; s < 128; ++s)
    acc += eb[s] * __builtin_nontemporal_load(base + (long long)s * KDIM);
  part[(((b << 4) + sc) << 10) + e] = acc;
}

// ---------------- kernel 6: reduce partials -> context [B, 1024]
__global__ void k_ctx_reduce(const float* __restrict__ part, float* __restrict__ out) {
  const int i = blockIdx.x * 256 + threadIdx.x;  // b*1024 + e
  const int b = i >> 10, e = i & 1023;
  float acc = 0.f;
#pragma unroll
  for (int sc = 0; sc < 16; ++sc) acc += part[(((b << 4) + sc) << 10) + e];
  out[i] = acc;
}

extern "C" void kernel_launch(void* const* d_in, const int* in_sizes, int n_in,
                              void* d_out, int out_size, void* d_ws, size_t ws_size,
                              hipStream_t stream) {
  const float* enc    = (const float*)d_in[0];
  const float* hidden = (const float*)d_in[1];
  const int*   lens   = (const int*)d_in[2];
  const float* attn_w = (const float*)d_in[3];
  const float* attn_b = (const float*)d_in[4];
  const float* v_w    = (const float*)d_in[5];
  const float* v_b    = (const float*)d_in[6];
  float* out = (float*)d_out;

  char* ws = (char*)d_ws;
  float*  hb     = (float*)(ws);                       //  65536 B
  float*  scores = (float*)(ws + 65536);               // 262144 B
  float*  energy = (float*)(ws + 327680);              // 262144 B
  float*  part   = (float*)(ws + 589824);              // 2097152 B
  __bf16* wbf    = (__bf16*)(ws + 2686976);            // 1048576 B  (~3.6 MB total)

  k_prep_hb <<<64, 256, 0, stream>>>(hidden, attn_w, attn_b, hb);
  k_prep_wbf<<<512, 256, 0, stream>>>(attn_w, wbf);
  k_scores  <<<(BB * SS) / TM, 128, 0, stream>>>(enc, wbf, hb, v_w, v_b, scores);
  k_softmax <<<BB, 256, 0, stream>>>(scores, lens, energy);
  dim3 gctx(4, 16, 32);
  k_ctx_part<<<gctx, 256, 0, stream>>>(enc, energy, part);
  k_ctx_reduce<<<128, 256, 0, stream>>>(part, out);
}
